// LocalSelfAttention_33449205301852
// MI455X (gfx1250) — matrix operand order repaired and checked by reading, not compiled, not run on hardware
//
#include <hip/hip_runtime.h>

typedef __attribute__((ext_vector_type(16))) __bf16 v16bf;
typedef __attribute__((ext_vector_type(8)))  float  v8f;

union BF16x16 {
    uint4 q[2];
    unsigned short u[16];
    v16bf v;
};

__device__ __forceinline__ unsigned short f2bf(float f) {
    unsigned u = __float_as_uint(f);
    unsigned r = u + 0x7fffu + ((u >> 16) & 1u);   // round-to-nearest-even
    return (unsigned short)(r >> 16);
}

__device__ __forceinline__ v8f wmma_bf16(v16bf a, v16bf b, v8f c) {
    return __builtin_amdgcn_wmma_f32_16x16x32_bf16(
        false, a, false, b, (short)0, c, false, false);
}

// A-frag (16x32 bf16): lane holds row (lane&15); elems 0-7 = K[kb..kb+7], 8-15 = K[kb+16..kb+23]
__device__ __forceinline__ v16bf load_fragA_bf16(const unsigned short* p) {
    BF16x16 t;
    t.q[0] = *(const uint4*)(p);
    t.q[1] = *(const uint4*)(p + 16);
    return t.v;
}

// B-frag (32x16 bf16): lane holds col (lane&15); elems 0-7 = K[kb..kb+7], 8-15 = K[kb+16..kb+23]
__device__ __forceinline__ v16bf load_fragB_bf16(const unsigned short* p) {
    BF16x16 t;
    t.q[0] = *(const uint4*)(p);
    t.q[1] = *(const uint4*)(p + 16);
    return t.v;
}

#define HID   1024
#define HEADS 16
#define HDIM  64
#define NTOK  4096
#define NBAT  8
#define WSZ   64          // window (bucket) size
#define KWIN  192         // 3 * WSZ keys per window
#define LSTR  200         // LDS row stride in bf16 elems (mult of 8 -> 16B aligned rows)

// ---------------------------------------------------------------------------
// Kernel 1a: convert Wq|Wk|Wv (fp32, row-major [1024,1024]) -> bf16
// ---------------------------------------------------------------------------
__global__ __launch_bounds__(256) void convert_w_kernel(
    const float* __restrict__ Wq, const float* __restrict__ Wk,
    const float* __restrict__ Wv, unsigned int* __restrict__ Wb /* packed 2xbf16 */) {
    const size_t PAIRS = (size_t)HID * HID / 2;           // per matrix
    size_t i = (size_t)blockIdx.x * blockDim.x + threadIdx.x;
    if (i >= 3 * PAIRS) return;
    int z = (int)(i / PAIRS);
    size_t j = i - (size_t)z * PAIRS;
    const float* src = (z == 0) ? Wq : (z == 1) ? Wk : Wv;
    float2 f = *(const float2*)(src + j * 2);
    Wb[i] = (unsigned)f2bf(f.x) | ((unsigned)f2bf(f.y) << 16);
}

// ---------------------------------------------------------------------------
// Kernel 1b: convert activations x (fp32 [32768,1024]) -> bf16 (memory-bound)
// ---------------------------------------------------------------------------
__global__ __launch_bounds__(256) void convert_x_kernel(
    const float* __restrict__ x, uint2* __restrict__ Xb /* 4x bf16 per uint2 */) {
    const size_t QUADS = (size_t)NBAT * NTOK * HID / 4;
    size_t i = (size_t)blockIdx.x * blockDim.x + threadIdx.x;
    if (i >= QUADS) return;
    float4 f = *(const float4*)(x + i * 4);
    uint2 o;
    o.x = (unsigned)f2bf(f.x) | ((unsigned)f2bf(f.y) << 16);
    o.y = (unsigned)f2bf(f.z) | ((unsigned)f2bf(f.w) << 16);
    Xb[i] = o;
}

// ---------------------------------------------------------------------------
// Kernel 2: QKV projection GEMM.  Y = x @ W^T + b  (bf16 out, [B,H,N,D] layout)
// grid = (N/128, M/256, 3), block = 256 (8 waves), wave tile 64x64
// Inner loop: pure b128 loads + 16 WMMAs (no conversion VALU).
// ---------------------------------------------------------------------------
__global__ __launch_bounds__(256) void qkv_gemm_kernel(
    const unsigned short* __restrict__ Xb, const unsigned short* __restrict__ Wb,
    const float* __restrict__ bq, const float* __restrict__ bk, const float* __restrict__ bv,
    unsigned short* __restrict__ qb, unsigned short* __restrict__ kb,
    unsigned short* __restrict__ vb) {
    const int z    = blockIdx.z;
    const unsigned short* Wm   = Wb + (size_t)z * HID * HID;
    const float*          bias = (z == 0) ? bq : (z == 1) ? bk : bv;
    unsigned short*       out  = (z == 0) ? qb : (z == 1) ? kb : vb;
    const float           scale = (z == 0) ? 0.125f : 1.0f;   // HEAD_DIM^-0.5

    const int tid  = threadIdx.x;
    const int lane = tid & 31;
    const int wid  = tid >> 5;
    const int col  = lane & 15;
    const int half = lane >> 4;

    const int m0 = blockIdx.y * 256 + (wid & 3) * 64;  // wave M base
    const int n0 = blockIdx.x * 128 + (wid >> 2) * 64; // wave N base

    v8f acc[4][4];
#pragma unroll
    for (int mf = 0; mf < 4; ++mf)
#pragma unroll
        for (int nf = 0; nf < 4; ++nf) acc[mf][nf] = (v8f)0.0f;

    const unsigned short* arow0 = Xb + (size_t)(m0 + col) * HID + half * 8;
    const unsigned short* brow0 = Wm + (size_t)(n0 + col) * HID + half * 8;

    for (int kk = 0; kk < HID; kk += 32) {
        v16bf a[4];
#pragma unroll
        for (int mf = 0; mf < 4; ++mf)
            a[mf] = load_fragA_bf16(arow0 + (size_t)mf * 16 * HID + kk);
#pragma unroll
        for (int nf = 0; nf < 4; ++nf) {
            v16bf bfrag = load_fragB_bf16(brow0 + (size_t)nf * 16 * HID + kk);
#pragma unroll
            for (int mf = 0; mf < 4; ++mf)
                acc[mf][nf] = wmma_bf16(a[mf], bfrag, acc[mf][nf]);
        }
    }

    // Epilogue: +bias, (*0.125 for q), bf16 store into [B,H,N,D]
#pragma unroll
    for (int nf = 0; nf < 4; ++nf) {
        const int c  = n0 + nf * 16 + col;
        const float bi = bias[c];
        const int hh = c >> 6, d = c & 63;
#pragma unroll
        for (int mf = 0; mf < 4; ++mf) {
#pragma unroll
            for (int vr = 0; vr < 8; ++vr) {
                const int m  = m0 + mf * 16 + vr + 8 * half;
                const int bbi = m >> 12;        // / 4096
                const int n   = m & 4095;
                const float y = (acc[mf][nf][vr] + bi) * scale;
                out[((size_t)(bbi * HEADS + hh) * NTOK + n) * HDIM + d] = f2bf(y);
            }
        }
    }
}

// ---------------------------------------------------------------------------
// Kernel 3: local attention. grid = (nw=64, H=16, B=8), block = 128 (4 waves)
// wave handles 16 query rows. scores [16,192] via WMMA, masked softmax,
// P -> LDS bf16, V staged transposed in LDS, O = P @ Vwin via WMMA.
// ---------------------------------------------------------------------------
__global__ __launch_bounds__(128) void local_attn_kernel(
    const unsigned short* __restrict__ qb, const unsigned short* __restrict__ kb,
    const unsigned short* __restrict__ vb, const unsigned char* __restrict__ amask,
    float* __restrict__ out) {
    const int w = blockIdx.x, h = blockIdx.y, b = blockIdx.z;
    const int tid  = threadIdx.x;
    const int lane = tid & 31;
    const int wid  = tid >> 5;
    const int col  = lane & 15;
    const int half = lane >> 4;

    __shared__ unsigned short Vt[HDIM * LSTR];   // transposed V window [d][key]
    __shared__ unsigned short Pt[WSZ * LSTR];    // softmax probs bf16 [row][key]

    const int    kfirst = (w - 1) * WSZ;                         // first key token (may be <0)
    const size_t bh     = ((size_t)b * HEADS + h) * NTOK * HDIM; // base of (b,h) plane
    const unsigned char* mrow_mask = amask + (size_t)b * NTOK;

    // ---- Stage V window transposed into LDS (zeros for padded keys) ----
    for (int idx = tid; idx < HDIM * (KWIN / 2); idx += 128) {
        const int d  = idx / (KWIN / 2);
        const int jp = idx - d * (KWIN / 2);
        const int j  = jp * 2;
        const int t0 = kfirst + j, t1 = t0 + 1;
        unsigned v0 = (t0 >= 0 && t0 < NTOK) ? (unsigned)vb[bh + (size_t)t0 * HDIM + d] : 0u;
        unsigned v1 = (t1 >= 0 && t1 < NTOK) ? (unsigned)vb[bh + (size_t)t1 * HDIM + d] : 0u;
        *(unsigned*)&Vt[d * LSTR + j] = v0 | (v1 << 16);
    }

    // ---- Q fragments (rows wid*16 .. +15), K=64 -> 2 k-steps ----
    const int mrow = wid * 16 + col;
    const int qtok = w * WSZ + mrow;
    const unsigned short* qrow = qb + bh + (size_t)qtok * HDIM;
    v16bf aq0 = load_fragA_bf16(qrow + 0  + half * 8);
    v16bf aq1 = load_fragA_bf16(qrow + 32 + half * 8);

    // ---- S = Q @ K^T  (12 tiles of 16 keys) ----
    v8f acc[12];
    bool kvalid[12];
#pragma unroll
    for (int nt = 0; nt < 12; ++nt) {
        const int kt  = kfirst + nt * 16 + col;
        const bool ir = (kt >= 0) && (kt < NTOK);
        kvalid[nt] = ir && (mrow_mask[ir ? kt : 0] != 0);
        const int ktc = ir ? kt : 0;
        const unsigned short* krow = kb + bh + (size_t)ktc * HDIM;
        v16bf b0 = load_fragB_bf16(krow + 0  + half * 8);
        v16bf b1 = load_fragB_bf16(krow + 32 + half * 8);
        v8f c = (v8f)0.0f;
        c = wmma_bf16(aq0, b0, c);
        c = wmma_bf16(aq1, b1, c);
        acc[nt] = c;
    }

    // ---- masked softmax per row (rows split across 16-lane halves) ----
#pragma unroll
    for (int vr = 0; vr < 8; ++vr) {
        const int  mr = wid * 16 + vr + 8 * half;
        const bool mq = mrow_mask[w * WSZ + mr] != 0;
        float sv[12];
        float mx = -3.0e38f;
#pragma unroll
        for (int nt = 0; nt < 12; ++nt) {
            float s = acc[nt][vr];
            s = (mq && kvalid[nt]) ? s : -1.0e9f;
            sv[nt] = s;
            mx = fmaxf(mx, s);
        }
#pragma unroll
        for (int off = 1; off < 16; off <<= 1)
            mx = fmaxf(mx, __shfl_xor(mx, off, 32));
        float sum = 0.0f;
#pragma unroll
        for (int nt = 0; nt < 12; ++nt) {
            float p = __expf(sv[nt] - mx);
            sv[nt] = p;
            sum += p;
        }
#pragma unroll
        for (int off = 1; off < 16; off <<= 1)
            sum += __shfl_xor(sum, off, 32);
        const float r = 1.0f / sum;
#pragma unroll
        for (int nt = 0; nt < 12; ++nt)
            Pt[mr * LSTR + nt * 16 + col] = f2bf(sv[nt] * r);
    }

    __syncthreads();

    // ---- O = P @ Vwin  (K = 192 -> 6 k-steps; 4 d-tiles) ----
    v8f oacc[4];
#pragma unroll
    for (int nt = 0; nt < 4; ++nt) oacc[nt] = (v8f)0.0f;

    const unsigned short* prow = Pt + (size_t)(wid * 16 + col) * LSTR;
#pragma unroll
    for (int k0 = 0; k0 < KWIN; k0 += 32) {
        v16bf a = load_fragA_bf16(prow + k0 + half * 8);
#pragma unroll
        for (int nt = 0; nt < 4; ++nt) {
            const unsigned short* vrow = Vt + (size_t)(nt * 16 + col) * LSTR;
            v16bf bfrag = load_fragB_bf16(vrow + k0 + half * 8);
            oacc[nt] = wmma_bf16(a, bfrag, oacc[nt]);
        }
    }

    // ---- write O to [B, N, H*D] fp32 ----
    const size_t obase = (size_t)b * NTOK * HID + (size_t)h * HDIM;
#pragma unroll
    for (int nt = 0; nt < 4; ++nt) {
#pragma unroll
        for (int vr = 0; vr < 8; ++vr) {
            const int mr  = wid * 16 + vr + 8 * half;
            const int tok = w * WSZ + mr;
            out[obase + (size_t)tok * HID + nt * 16 + col] = oacc[nt][vr];
        }
    }
}

// ---------------------------------------------------------------------------
extern "C" void kernel_launch(void* const* d_in, const int* in_sizes, int n_in,
                              void* d_out, int out_size, void* d_ws, size_t ws_size,
                              hipStream_t stream) {
    const float*         x     = (const float*)d_in[0];
    const unsigned char* amask = (const unsigned char*)d_in[1];
    const float* Wq = (const float*)d_in[2];
    const float* bq = (const float*)d_in[3];
    const float* Wk = (const float*)d_in[4];
    const float* bk = (const float*)d_in[5];
    const float* Wv = (const float*)d_in[6];
    const float* bv = (const float*)d_in[7];
    float* out = (float*)d_out;

    // workspace layout (bf16): Wb[3*1024*1024] | Xb[32768*1024] | qb | kb | vb
    unsigned short* Wb = (unsigned short*)d_ws;
    const size_t WB_ELEMS  = (size_t)3 * HID * HID;
    const size_t X_ELEMS   = (size_t)NBAT * NTOK * HID;
    const size_t QKV_ELEMS = (size_t)NBAT * HEADS * NTOK * HDIM;
    unsigned short* Xb = Wb + WB_ELEMS;
    unsigned short* qb = Xb + X_ELEMS;
    unsigned short* kb = qb + QKV_ELEMS;
    unsigned short* vb = kb + QKV_ELEMS;

    // 1a) weights fp32 -> bf16
    {
        const size_t pairs = (size_t)3 * HID * HID / 2;
        dim3 grid((unsigned)((pairs + 255) / 256));
        convert_w_kernel<<<grid, 256, 0, stream>>>(Wq, Wk, Wv, (unsigned int*)Wb);
    }
    // 1b) activations fp32 -> bf16
    {
        const size_t quads = X_ELEMS / 4;
        dim3 grid((unsigned)((quads + 255) / 256));
        convert_x_kernel<<<grid, 256, 0, stream>>>(x, (uint2*)Xb);
    }
    // 2) QKV projection GEMMs (z = q,k,v)
    {
        dim3 grid(HID / 128, (NBAT * NTOK) / 256, 3);
        qkv_gemm_kernel<<<grid, 256, 0, stream>>>(Xb, Wb, bq, bk, bv, qb, kb, vb);
    }
    // 3) windowed attention
    {
        dim3 grid(NTOK / WSZ, HEADS, NBAT);
        local_attn_kernel<<<grid, 128, 0, stream>>>(qb, kb, vb, amask, out);
    }
}
